// Ece_26422638805243
// MI455X (gfx1250) — compile-verified, matrix operand order checked by reading, not run on hardware
//
#include <hip/hip_runtime.h>
#include <hip/hip_bf16.h>

typedef float     v2f  __attribute__((ext_vector_type(2)));
typedef float     v8f  __attribute__((ext_vector_type(8)));
typedef _Float16  v16h __attribute__((ext_vector_type(16)));

#define NCLS   100     // classes (C)
#define NBINS  15
#define RPB    128     // rows per block in softmax kernel
#define TPB1   256
#define CSTR   105     // padded LDS row stride (odd mod 64 -> conflict free)

// ---------------------------------------------------------------------------
// K1: softmax + transpose.  Each block: 128 rows.  Row-sum of exp() done with
// V_WMMA_F32_16X16X4_F32 (B = ones), 16 rows per wave, 25 K-chunks of 4.
// ---------------------------------------------------------------------------
__global__ __launch_bounds__(TPB1)
void k_softmaxT(const float* __restrict__ logits, float* __restrict__ probsT, int N) {
  __shared__ float tile[RPB * CSTR];
  __shared__ float rmax[RPB];
  __shared__ float rsum[RPB];
  __shared__ float rinv[RPB];

  const int tid  = threadIdx.x;
  const int row0 = blockIdx.x * RPB;

  // load 128x100 tile (coalesced)
  for (int i = tid; i < RPB * NCLS; i += TPB1) {
    int r = i / NCLS, c = i - r * NCLS;
    int row = row0 + r;
    tile[r * CSTR + c] = (row < N) ? logits[(size_t)row * NCLS + c] : 0.0f;
  }
  __syncthreads();

  // per-row max (threads 0..127, conflict-free: stride 105)
  if (tid < RPB) {
    float m = tile[tid * CSTR];
    for (int c = 1; c < NCLS; ++c) m = fmaxf(m, tile[tid * CSTR + c]);
    rmax[tid] = m;
  }
  __syncthreads();

  // exponentiate in place
  for (int i = tid; i < RPB * NCLS; i += TPB1) {
    int r = i / NCLS, c = i - r * NCLS;
    tile[r * CSTR + c] = __expf(tile[r * CSTR + c] - rmax[r]);
  }
  __syncthreads();

  // WMMA row sums: wave w owns rows 16w..16w+15
  const int lane = tid & 31;
  const int wave = tid >> 5;
  const int half = lane >> 4;     // A-matrix 16x4 f32: lanes 0-15 K=0,1; 16-31 K=2,3
  const int mrow = lane & 15;
  const int rb   = wave * 16;

#if __has_builtin(__builtin_amdgcn_wmma_f32_16x16x4_f32)
  v8f acc = {};
  v2f bones; bones.x = 1.0f; bones.y = 1.0f;
  for (int kb = 0; kb + 3 < NCLS; kb += 4) {          // 100 % 4 == 0 -> covers all
    v2f a;
    a.x = tile[(rb + mrow) * CSTR + kb + 2 * half + 0];
    a.y = tile[(rb + mrow) * CSTR + kb + 2 * half + 1];
    acc = __builtin_amdgcn_wmma_f32_16x16x4_f32(false, a, false, bones,
                                                (short)0, acc, false, false);
  }
#else
  // fallback: codegen-confirmed f16 WMMA 16x16x32 (B = ones)
  v8f acc = {};
  v16h bh;
  #pragma unroll
  for (int e = 0; e < 16; ++e) bh[e] = (_Float16)1.0f;
  for (int kb = 0; kb < NCLS; kb += 32) {
    v16h ah;
    #pragma unroll
    for (int v = 0; v < 8; ++v) {
      // 16-bit A 16x32 layout: V0..3 -> K=0..7 (+8 for upper half-wave), V4..7 -> K=16..23 (+8)
      int off = ((v < 4) ? (2 * v) : (2 * v + 8)) + (half ? 8 : 0) + kb;
      float x0 = (off     < NCLS) ? tile[(rb + mrow) * CSTR + off]     : 0.0f;
      float x1 = (off + 1 < NCLS) ? tile[(rb + mrow) * CSTR + off + 1] : 0.0f;
      ah[2 * v]     = (_Float16)x0;
      ah[2 * v + 1] = (_Float16)x1;
    }
    acc = __builtin_amdgcn_wmma_f32_16x16x32_f16(false, ah, false, bh,
                                                 (short)0, acc, false, false);
  }
#endif
  // D layout (16x16 f32): VGPR j, lanes 0-15 -> M=j, lanes 16-31 -> M=j+8.
  // Row sums are replicated along N, so lane 0 / lane 16 each hold 8 row sums.
  if (lane == 0) {
    #pragma unroll
    for (int j = 0; j < 8; ++j) rsum[rb + j] = acc[j];
  }
  if (lane == 16) {
    #pragma unroll
    for (int j = 0; j < 8; ++j) rsum[rb + 8 + j] = acc[j];
  }
  __syncthreads();

  if (tid < RPB) rinv[tid] = 1.0f / rsum[tid];
  __syncthreads();

  // transposed write: consecutive tid -> consecutive rows -> coalesced global
  for (int i = tid; i < NCLS * RPB; i += TPB1) {
    int c = i >> 7, r = i & (RPB - 1);
    int row = row0 + r;
    if (row < N) probsT[(size_t)c * N + row] = tile[r * CSTR + c] * rinv[r];
  }
}

// ---------------------------------------------------------------------------
// K2: per-class exact order-statistic selection via 3-level radix histograms.
// One block (1024 threads) per class; dynamic LDS = 16384 + 8192 + 1024 u32.
// Produces the 16 interpolated adaptive bin bounds.
// ---------------------------------------------------------------------------
__global__ __launch_bounds__(1024)
void k_select(const float* __restrict__ probsT, float* __restrict__ bounds, int N) {
  extern __shared__ unsigned char dynsmem[];
  unsigned* hist1 = (unsigned*)dynsmem;      // 16384: exponent+hi-mantissa buckets
  unsigned* hist2 = hist1 + 16384;           // 32 slots x 256 byte-buckets
  unsigned* part  = hist2 + 32 * 256;        // 1024 scan partials

  __shared__ int      s_rank[32];
  __shared__ float    s_frac[16];
  __shared__ unsigned s_bkt[32];
  __shared__ unsigned s_rem[32];
  __shared__ unsigned s_pref[32];
  __shared__ float    s_val[32];

  const int c   = blockIdx.x;
  const int tid = threadIdx.x;
  const int NT  = blockDim.x;                // 1024
  const float* col = probsT + (size_t)c * N;

  // ---- pass 1: histogram of key[30:16] (positive floats: bits order values)
  for (int i = tid; i < 16384; i += NT) hist1[i] = 0;
  __syncthreads();
  for (int i = tid; i < N; i += NT) {
    __builtin_prefetch(&col[i + NT * 8], 0, 1);
    unsigned key = __float_as_uint(col[i]);
    unsigned b = key >> 16; if (b > 16383u) b = 16383u;
    atomicAdd(&hist1[b], 1u);
  }
  __syncthreads();

  // ---- exclusive prefix scan over 16384 buckets (1024 threads x 16)
  unsigned loc[16]; unsigned mysum = 0;
  #pragma unroll
  for (int j = 0; j < 16; ++j) { loc[j] = hist1[tid * 16 + j]; mysum += loc[j]; }
  part[tid] = mysum;
  __syncthreads();
  for (int off = 1; off < 1024; off <<= 1) {
    unsigned add = (tid >= off) ? part[tid - off] : 0u;
    __syncthreads();
    part[tid] += add;
    __syncthreads();
  }
  unsigned run = part[tid] - mysum;          // exclusive base of my chunk
  #pragma unroll
  for (int j = 0; j < 16; ++j) { unsigned v = loc[j]; hist1[tid * 16 + j] = run; run += v; }
  __syncthreads();

  // ---- needed ranks: j_k = floor(k*N/15) and j_k+1 (jnp.interp on sorted data)
  const float step = (float)N / (float)NBINS;
  if (tid < 16) {
    float pos = step * (float)tid;
    float jf  = floorf(pos);
    int j  = (int)jf; if (j > N - 1) j = N - 1;
    int j2 = j + 1;   if (j2 > N - 1) j2 = N - 1;
    s_rank[2 * tid]     = j;
    s_rank[2 * tid + 1] = j2;
    s_frac[tid] = (j2 > j) ? (pos - jf) : 0.0f;
  }
  __syncthreads();
  if (tid < 32) {                             // bucket containing each rank
    int r = s_rank[tid];
    int lo = 0, hi = 16383;
    while (lo < hi) { int mid = (lo + hi + 1) >> 1; if ((int)hist1[mid] <= r) lo = mid; else hi = mid - 1; }
    s_bkt[tid] = (unsigned)lo;
    s_rem[tid] = (unsigned)(r - (int)hist1[lo]);
  }
  __syncthreads();

  // ---- pass 2: per-slot histogram of key[15:8] within each needed bucket
  for (int i = tid; i < 32 * 256; i += NT) hist2[i] = 0;
  __syncthreads();
  unsigned tb[32];
  #pragma unroll
  for (int s = 0; s < 32; ++s) tb[s] = s_bkt[s];
  for (int i = tid; i < N; i += NT) {
    __builtin_prefetch(&col[i + NT * 8], 0, 1);
    unsigned key = __float_as_uint(col[i]);
    unsigned hb  = key >> 16;
    unsigned mid = (key >> 8) & 0xFFu;
    #pragma unroll
    for (int s = 0; s < 32; ++s)
      if (hb == tb[s]) atomicAdd(&hist2[s * 256 + mid], 1u);
  }
  __syncthreads();
  if (tid < 32) {
    unsigned rem = s_rem[tid], acc = 0, sel = 255;
    for (int j = 0; j < 256; ++j) {
      unsigned cj = hist2[tid * 256 + j];
      if (acc + cj > rem) { sel = (unsigned)j; break; }
      acc += cj;
    }
    s_pref[tid] = (s_bkt[tid] << 8) | sel;    // 24-bit key prefix
    s_rem[tid]  = rem - acc;
  }
  __syncthreads();

  // ---- pass 3: per-slot histogram of key[7:0] within 24-bit prefix
  for (int i = tid; i < 32 * 256; i += NT) hist2[i] = 0;
  __syncthreads();
  unsigned tp[32];
  #pragma unroll
  for (int s = 0; s < 32; ++s) tp[s] = s_pref[s];
  for (int i = tid; i < N; i += NT) {
    __builtin_prefetch(&col[i + NT * 8], 0, 1);
    unsigned key = __float_as_uint(col[i]);
    unsigned hp  = key >> 8;
    unsigned lo8 = key & 0xFFu;
    #pragma unroll
    for (int s = 0; s < 32; ++s)
      if (hp == tp[s]) atomicAdd(&hist2[s * 256 + lo8], 1u);
  }
  __syncthreads();
  if (tid < 32) {
    unsigned rem = s_rem[tid], acc = 0, sel = 255;
    for (int j = 0; j < 256; ++j) {
      unsigned cj = hist2[tid * 256 + j];
      if (acc + cj > rem) { sel = (unsigned)j; break; }
      acc += cj;
    }
    s_val[tid] = __uint_as_float((s_pref[tid] << 8) | sel);  // exact srt[rank]
  }
  __syncthreads();

  if (tid < 16) {
    float vlo = s_val[2 * tid], vhi = s_val[2 * tid + 1];
    bounds[c * 16 + tid] = vlo + s_frac[tid] * (vhi - vlo);
  }
}

// ---------------------------------------------------------------------------
// K3: bin assignment + per-(class,bin,label) {count, Sconf, Sconf2} stats.
// grid = C * SPLITS blocks; per-wave LDS accumulators, then global atomics.
// ---------------------------------------------------------------------------
__global__ __launch_bounds__(256)
void k_bins(const float* __restrict__ probsT, const int* __restrict__ labels,
            const float* __restrict__ bounds, float* __restrict__ stats,
            int N, int S) {
  __shared__ float    sb[16];
  __shared__ unsigned wc[8][32];
  __shared__ float    w1[8][32];
  __shared__ float    w2[8][32];

  const int bid = blockIdx.x;
  const int c   = bid / S;
  const int sp  = bid - c * S;
  const int tid = threadIdx.x;
  const int wave = tid >> 5;

  if (tid < 16) sb[tid] = bounds[c * 16 + tid];
  for (int i = tid; i < 8 * 32; i += blockDim.x) {
    ((unsigned*)wc)[i] = 0u; ((float*)w1)[i] = 0.0f; ((float*)w2)[i] = 0.0f;
  }
  __syncthreads();

  const int chunk = (N + S - 1) / S;
  const int i0 = sp * chunk;
  const int i1 = min(N, i0 + chunk);
  const float* col = probsT + (size_t)c * N;

  for (int i = i0 + tid; i < i1; i += blockDim.x) {
    __builtin_prefetch(&col[i + blockDim.x * 8], 0, 1);
    float conf = col[i];
    int cnt = 0;                               // searchsorted(bounds, conf, 'left')
    #pragma unroll
    for (int j = 0; j < 16; ++j) cnt += (sb[j] < conf) ? 1 : 0;
    int bin = cnt - 1;
    if (bin >= 0 && bin < NBINS) {
      int lab  = (labels[i] == c) ? 1 : 0;
      int slot = bin * 2 + lab;
      atomicAdd(&wc[wave][slot], 1u);
      atomicAdd(&w1[wave][slot], conf);
      atomicAdd(&w2[wave][slot], conf * conf);
    }
  }
  __syncthreads();

  if (tid < 32) {
    unsigned tc = 0; float t1 = 0.0f, t2 = 0.0f;
    #pragma unroll
    for (int w = 0; w < 8; ++w) { tc += wc[w][tid]; t1 += w1[w][tid]; t2 += w2[w][tid]; }
    int bin = tid >> 1, lab = tid & 1;
    if (bin < NBINS && tc > 0) {
      float* dst = stats + (((size_t)c * 16 + bin) * 2 + lab) * 3;
      atomicAdd(dst + 0, (float)tc);
      atomicAdd(dst + 1, t1);
      atomicAdd(dst + 2, t2);
    }
  }
}

// ---------------------------------------------------------------------------
// K4: closed-form leave-one-out contributions from sufficient statistics.
// sum over lab=0: S2 - 2*c0*S1 + n0*c0^2 with c0 = n1/(nb-1); lab=1 analogous.
// ---------------------------------------------------------------------------
__global__ __launch_bounds__(128)
void k_final(const float* __restrict__ stats, float* __restrict__ out, int N) {
  __shared__ float red[128];
  const int t = threadIdx.x;
  float acc = 0.0f;
  if (t < NCLS) {
    for (int b = 0; b < NBINS; ++b) {
      const float* p = stats + (((size_t)t * 16 + b) * 2) * 3;
      float n0 = p[0], s10 = p[1], s20 = p[2];
      float n1 = p[3], s11 = p[4], s21 = p[5];
      float nb = n0 + n1;
      if (nb > 1.0f) {
        float d  = nb - 1.0f;
        float c0 = n1 / d;
        float c1 = (n1 - 1.0f) / d;
        acc += (s20 - 2.0f * c0 * s10 + n0 * c0 * c0)
             + (s21 - 2.0f * c1 * s11 + n1 * c1 * c1);
      }
    }
  }
  red[t] = acc;
  __syncthreads();
  for (int off = 64; off > 0; off >>= 1) {
    if (t < off) red[t] += red[t + off];
    __syncthreads();
  }
  if (t == 0) out[0] = red[0] / ((float)N * (float)NCLS);
}

// ---------------------------------------------------------------------------
extern "C" void kernel_launch(void* const* d_in, const int* in_sizes, int n_in,
                              void* d_out, int out_size, void* d_ws, size_t ws_size,
                              hipStream_t stream) {
  const float* logits = (const float*)d_in[0];
  const int*   labels = (const int*)d_in[1];
  float*       out    = (float*)d_out;

  const int N = in_sizes[1];                       // 500000
  // C fixed at NCLS=100 per reference shapes.

  // workspace layout
  size_t probsBytes = (size_t)N * NCLS * sizeof(float);
  float* probsT = (float*)d_ws;
  float* bounds = (float*)((char*)d_ws + probsBytes);
  float* stats  = bounds + (size_t)NCLS * 16;       // [C][16][2][3] f32

  (void)hipMemsetAsync(stats, 0, (size_t)NCLS * 16 * 2 * 3 * sizeof(float), stream);

  const int grid1 = (N + RPB - 1) / RPB;
  k_softmaxT<<<grid1, TPB1, 0, stream>>>(logits, probsT, N);

  const size_t dynLds = (16384 + 32 * 256 + 1024) * sizeof(unsigned);  // 102400 B
  k_select<<<NCLS, 1024, dynLds, stream>>>(probsT, bounds, N);

  const int SPLITS = 8;
  k_bins<<<NCLS * SPLITS, 256, 0, stream>>>(probsT, labels, bounds, stats, N, SPLITS);

  k_final<<<1, 128, 0, stream>>>(stats, out, N);
}
